// MiniAttention_53257594470777
// MI455X (gfx1250) — matrix operand order repaired and checked by reading, not compiled, not run on hardware
//
#include <hip/hip_runtime.h>

#define B_    4
#define L_    2048
#define D_    1024
#define H_    16
#define DK_   64
#define MTOK  (B_ * L_)        // 8192 tokens
#define N_QKV (3 * D_)         // 3072
// (1/sqrt(d_k)) * log2(e), folded into Q so softmax uses exp2
#define QSCALE 0.18033688011112042f

typedef __attribute__((ext_vector_type(16))) __bf16 v16bf;
typedef __attribute__((ext_vector_type(8)))  float  v8f;
typedef __attribute__((ext_vector_type(4)))  unsigned u32x4;
typedef __attribute__((ext_vector_type(8)))  int      i32x8;
typedef __attribute__((ext_vector_type(4)))  int      i32x4;

#if __has_builtin(__builtin_amdgcn_tensor_load_to_lds) && \
    __has_builtin(__builtin_amdgcn_s_wait_tensorcnt)
#define HAVE_TDM 1
#else
#define HAVE_TDM 0
#endif

union Frag {
  v16bf   v;
  unsigned u[8];
  __bf16  h[16];
};

__device__ __forceinline__ v8f wmma_bf16(const Frag& a, const Frag& b, v8f c) {
  // D = A(16x32 bf16) x B(32x16 bf16) + C(16x16 f32)
  return __builtin_amdgcn_wmma_f32_16x16x32_bf16(
      /*neg_a=*/false, a.v, /*neg_b=*/false, b.v,
      /*c_mod=*/(short)0, c, /*reuse_a=*/false, /*reuse_b=*/false);
}

#if HAVE_TDM
// Issue one TDM 2D tile load: global (tile_d1 rows x tile_d0 bf16 elems,
// row stride stride0 elems) -> LDS at lds_off, rows packed contiguously.
// Whole-wave instruction (EXEC ignored); caller gates to one wave per block.
__device__ __forceinline__ void tdm_load_2d_bf16(unsigned lds_off, const void* gptr,
                                                 unsigned tile_d0, unsigned tile_d1,
                                                 unsigned stride0) {
  unsigned long long ga = (unsigned long long)(uintptr_t)gptr;
  u32x4 g0;
  g0[0] = 1u;                                           // count=1, user D#
  g0[1] = lds_off;                                      // lds_addr (bytes)
  g0[2] = (unsigned)(ga & 0xffffffffull);               // global_addr[31:0]
  g0[3] = (unsigned)((ga >> 32) & 0x1ffffffull)         // global_addr[56:32]
          | (2u << 30);                                 // type=2 ("image")
  i32x8 g1;
  g1[0] = (int)(1u << 16);                              // data_size=1 -> 2B
  g1[1] = (int)((tile_d0 & 0xffffu) << 16);             // tensor_dim0[15:0]
  g1[2] = (int)((tile_d0 >> 16) & 0xffffu)              // tensor_dim0[31:16]
          | (int)((tile_d1 & 0xffffu) << 16);           // tensor_dim1[15:0]
  g1[3] = (int)((tile_d1 >> 16) & 0xffffu)              // tensor_dim1[31:16]
          | (int)((tile_d0 & 0xffffu) << 16);           // tile_dim0
  g1[4] = (int)(tile_d1 & 0xffffu);                     // tile_dim1 (tile_dim2=0)
  g1[5] = (int)stride0;                                 // tensor_dim0_stride[31:0]
  g1[6] = 0;                                            // stride hi / dim1_stride lo
  g1[7] = 0;
  i32x4 z4 = {0, 0, 0, 0};
  i32x8 z8 = {0, 0, 0, 0, 0, 0, 0, 0};
  __builtin_amdgcn_tensor_load_to_lds(g0, g1, z4, z4, z8, 0);
}
#endif

// ---------------------------------------------------------------------------
// fp32 -> bf16 conversion, 4 elements per thread
// ---------------------------------------------------------------------------
__global__ __launch_bounds__(256)
void cvt_f32_bf16_x4(const float* __restrict__ src, __bf16* __restrict__ dst, int n4) {
  int i = blockIdx.x * 256 + threadIdx.x;
  if (i < n4) {
    float4 f = ((const float4*)src)[i];
    union { __bf16 h[4]; uint2 u; } o;
    o.h[0] = (__bf16)f.x; o.h[1] = (__bf16)f.y;
    o.h[2] = (__bf16)f.z; o.h[3] = (__bf16)f.w;
    ((uint2*)dst)[i] = o.u;
  }
}

// ---------------------------------------------------------------------------
// LayerNorm (one 1024-elem row per block) -> bf16
// ---------------------------------------------------------------------------
__global__ __launch_bounds__(256)
void layernorm_bf16(const float* __restrict__ x, const float* __restrict__ g,
                    const float* __restrict__ be, __bf16* __restrict__ xn) {
  const int row = blockIdx.x;
  const int tid = threadIdx.x;
  const float* xr = x + (size_t)row * D_;
  float4 f = ((const float4*)xr)[tid];
  float s = f.x + f.y + f.z + f.w;
  float q = f.x * f.x + f.y * f.y + f.z * f.z + f.w * f.w;
#pragma unroll
  for (int off = 16; off > 0; off >>= 1) {
    s += __shfl_down(s, off, 32);
    q += __shfl_down(q, off, 32);
  }
  __shared__ float ss[8], sq[8];
  const int wv = tid >> 5, lane = tid & 31;
  if (lane == 0) { ss[wv] = s; sq[wv] = q; }
  __syncthreads();
  float ts = 0.f, tq = 0.f;
#pragma unroll
  for (int i = 0; i < 8; ++i) { ts += ss[i]; tq += sq[i]; }
  const float mu   = ts * (1.0f / D_);
  const float var  = tq * (1.0f / D_) - mu * mu;
  const float rstd = rsqrtf(var + 1e-5f);
  float4 gg = ((const float4*)g)[tid];
  float4 bb = ((const float4*)be)[tid];
  union { __bf16 h[4]; uint2 u; } o;
  o.h[0] = (__bf16)((f.x - mu) * rstd * gg.x + bb.x);
  o.h[1] = (__bf16)((f.y - mu) * rstd * gg.y + bb.y);
  o.h[2] = (__bf16)((f.z - mu) * rstd * gg.z + bb.z);
  o.h[3] = (__bf16)((f.w - mu) * rstd * gg.w + bb.w);
  ((uint2*)(xn + (size_t)row * D_))[tid] = o.u;
}

// ---------------------------------------------------------------------------
// Tiled bf16 WMMA GEMM: C[M,N] = A[M,K] * B[K,N] (+ epilogues)
//  mode 0: QKV projection -> scatter q/k/v (bf16, [B,H,L,64]), Q pre-scaled
//  mode 1: FC projection  -> +bias +fp32 residual -> fp32 out
// Block tile 128x128, K-step 64, 8 waves (2x4) of 64x32 each.
// A tile staged by the Tensor Data Mover; B tile transposed manually.
// ---------------------------------------------------------------------------
__global__ __launch_bounds__(256)
void gemm_bf16_kernel(const __bf16* __restrict__ A, const __bf16* __restrict__ Bw,
                      const float* __restrict__ bias, int M, int N, int K, int mode,
                      __bf16* __restrict__ qbuf, __bf16* __restrict__ kbuf,
                      __bf16* __restrict__ vbuf,
                      const float* __restrict__ residual, float* __restrict__ outf) {
  __shared__ __align__(16) __bf16 sA[128 * 64];   // [m][k]
  __shared__ __align__(16) __bf16 sBT[128 * 64];  // [n][k]

  const int tid  = threadIdx.x;
  const int lane = tid & 31;
  const int wv   = tid >> 5;
  const int wm   = wv >> 2;   // 0..1 : 64-row slab
  const int wn   = wv & 3;    // 0..3 : 32-col slab
  const int hh   = lane >> 4; // lane half
  const int ml   = lane & 15;

  int koffA[8], koffB[8];
#pragma unroll
  for (int j = 0; j < 8; ++j) {
    koffA[j] = (j < 4) ? (hh * 8 + 2 * j) : (16 + hh * 8 + 2 * (j - 4));
    koffB[j] = (j < 4) ? (hh * 8 + 2 * j) : (16 + hh * 8 + 2 * (j - 4));
  }

  const int m0 = blockIdx.y * 128;
  const int n0 = blockIdx.x * 128;

  v8f c[4][2] = {};

  for (int k0 = 0; k0 < K; k0 += 64) {
    __syncthreads();
#if HAVE_TDM
    // A tile via Tensor Data Mover: 128 rows x 64 bf16, row stride K
    if (wv == 0)
      tdm_load_2d_bf16((unsigned)(uintptr_t)(void*)&sA[0],
                       &A[(size_t)m0 * K + k0], 64u, 128u, (unsigned)K);
#else
#pragma unroll
    for (int i = 0; i < 4; ++i) {
      int t = tid + i * 256;
      int r = t >> 3, c8 = t & 7;
      *(uint4*)&sA[r * 64 + c8 * 8] =
          *(const uint4*)&A[(size_t)(m0 + r) * K + k0 + c8 * 8];
    }
#endif
    // stage B tile transposed: global [k][n] -> LDS [n][k]
#pragma unroll
    for (int i = 0; i < 4; ++i) {
      int t = tid + i * 256;
      int r = t >> 4, c8 = t & 15;
      uint4 d = *(const uint4*)&Bw[(size_t)(k0 + r) * N + n0 + c8 * 8];
      const __bf16* dp = (const __bf16*)&d;
#pragma unroll
      for (int e = 0; e < 8; ++e) sBT[(c8 * 8 + e) * 64 + r] = dp[e];
      if (k0 + 64 < K)
        __builtin_prefetch(&Bw[(size_t)(k0 + 64 + r) * N + n0 + c8 * 8], 0, 1);
    }
#if HAVE_TDM
    if (wv == 0) __builtin_amdgcn_s_wait_tensorcnt(0);
#endif
    __syncthreads();

#pragma unroll
    for (int kk = 0; kk < 64; kk += 32) {
      Frag a[4];
#pragma unroll
      for (int i = 0; i < 4; ++i) {
        int row = wm * 64 + i * 16 + ml;
#pragma unroll
        for (int j = 0; j < 8; ++j)
          a[i].u[j] = *(const unsigned*)&sA[row * 64 + kk + koffA[j]];
      }
      Frag b[2];
#pragma unroll
      for (int jn = 0; jn < 2; ++jn) {
        int col = wn * 32 + jn * 16 + ml;
#pragma unroll
        for (int j = 0; j < 8; ++j)
          b[jn].u[j] = *(const unsigned*)&sBT[col * 64 + kk + koffB[j]];
      }
#pragma unroll
      for (int i = 0; i < 4; ++i)
#pragma unroll
        for (int jn = 0; jn < 2; ++jn)
          c[i][jn] = wmma_bf16(a[i], b[jn], c[i][jn]);
    }
  }

  // epilogue: C layout -> element (m = r + 8*hh, n = ml) per vgpr r
#pragma unroll
  for (int i = 0; i < 4; ++i) {
#pragma unroll
    for (int jn = 0; jn < 2; ++jn) {
      const int n  = n0 + wn * 32 + jn * 16 + ml;
      const float bv = bias[n];
#pragma unroll
      for (int r = 0; r < 8; ++r) {
        const int m = m0 + wm * 64 + i * 16 + r + 8 * hh;
        float val = c[i][jn][r] + bv;
        if (mode == 0) {
          const int which = n >> 10;        // 0:q 1:k 2:v
          const int rem   = n & 1023;
          const int head  = rem >> 6;
          const int dk    = rem & 63;
          const int bb    = m >> 11;        // / L_
          const int ll    = m & 2047;
          const size_t idx =
              ((size_t)(bb * H_ + head) * L_ + ll) * DK_ + dk;
          if (which == 0)      qbuf[idx] = (__bf16)(val * QSCALE);
          else if (which == 1) kbuf[idx] = (__bf16)val;
          else                 vbuf[idx] = (__bf16)val;
        } else {
          const size_t idx = (size_t)m * N + n;
          outf[idx] = val + residual[idx];
        }
      }
    }
  }
}

// ---------------------------------------------------------------------------
// Flash attention: one (b,h) x 64 query rows per block; 4 waves x 16 rows.
// S = Q*K^T via WMMA, online softmax (exp2, scale pre-folded into Q),
// P -> LDS (C-layout -> A-layout), O += P*V via WMMA.
// K tile staged by TDM; V tile transposed manually.
// ---------------------------------------------------------------------------
__global__ __launch_bounds__(128)
void attn_kernel(const __bf16* __restrict__ qb, const __bf16* __restrict__ kb,
                 const __bf16* __restrict__ vb, __bf16* __restrict__ ob) {
  __shared__ __align__(16) __bf16 sK[64 * 64];       // [key][dk]
  __shared__ __align__(16) __bf16 sVT[64 * 64];      // [dk][key]
  __shared__ __align__(16) __bf16 sP[4 * 16 * 64];   // per-wave [m][key]

  const int tid  = threadIdx.x;
  const int lane = tid & 31;
  const int wv   = tid >> 5;   // 0..3
  const int hh   = lane >> 4;
  const int ml   = lane & 15;

  const int bh   = blockIdx.y;
  const int bI   = bh >> 4;
  const int head = bh & 15;

  const __bf16* qh = qb + (size_t)bh * L_ * DK_;
  const __bf16* kh = kb + (size_t)bh * L_ * DK_;
  const __bf16* vh = vb + (size_t)bh * L_ * DK_;

  int koffA[8], koffB[8];
#pragma unroll
  for (int j = 0; j < 8; ++j) {
    koffA[j] = (j < 4) ? (hh * 8 + 2 * j) : (16 + hh * 8 + 2 * (j - 4));
    koffB[j] = (j < 4) ? (hh * 8 + 2 * j) : (16 + hh * 8 + 2 * (j - 4));
  }

  const int q0 = blockIdx.x * 64 + wv * 16;

  // Q fragments (A operand), kept in registers for the whole kernel
  Frag qf[2];
#pragma unroll
  for (int kk2 = 0; kk2 < 2; ++kk2)
#pragma unroll
    for (int j = 0; j < 8; ++j)
      qf[kk2].u[j] =
          *(const unsigned*)&qh[(size_t)(q0 + ml) * DK_ + kk2 * 32 + koffA[j]];

  float mstate[8], lstate[8];
#pragma unroll
  for (int r = 0; r < 8; ++r) { mstate[r] = -1e30f; lstate[r] = 0.f; }
  v8f o[4] = {};

  for (int kt = 0; kt < L_ / 64; ++kt) {
    __syncthreads();
#if HAVE_TDM
    // K tile via TDM: 64 keys x 64 bf16, fully contiguous
    if (wv == 0)
      tdm_load_2d_bf16((unsigned)(uintptr_t)(void*)&sK[0],
                       &kh[(size_t)(kt * 64) * DK_], 64u, 64u, 64u);
    // V tile transposed -> [dk][key] (manual)
#pragma unroll
    for (int i = 0; i < 4; ++i) {
      int t = tid + i * 128;
      int key = t >> 3, c8 = t & 7;
      uint4 d = *(const uint4*)&vh[(size_t)(kt * 64 + key) * DK_ + c8 * 8];
      const __bf16* dp = (const __bf16*)&d;
#pragma unroll
      for (int e = 0; e < 8; ++e) sVT[(c8 * 8 + e) * 64 + key] = dp[e];
    }
    if (wv == 0) __builtin_amdgcn_s_wait_tensorcnt(0);
#else
#pragma unroll
    for (int i = 0; i < 4; ++i) {
      int t = tid + i * 128;
      int key = t >> 3, c8 = t & 7;
      *(uint4*)&sK[key * 64 + c8 * 8] =
          *(const uint4*)&kh[(size_t)(kt * 64 + key) * DK_ + c8 * 8];
      uint4 d = *(const uint4*)&vh[(size_t)(kt * 64 + key) * DK_ + c8 * 8];
      const __bf16* dp = (const __bf16*)&d;
#pragma unroll
      for (int e = 0; e < 8; ++e) sVT[(c8 * 8 + e) * 64 + key] = dp[e];
    }
#endif
    __syncthreads();

    // S = Q * K^T : 16 q-rows x 64 keys
    v8f s[4] = {};
#pragma unroll
    for (int jn = 0; jn < 4; ++jn) {
#pragma unroll
      for (int kk2 = 0; kk2 < 2; ++kk2) {
        Frag bf_;
#pragma unroll
        for (int j = 0; j < 8; ++j)
          bf_.u[j] =
              *(const unsigned*)&sK[(jn * 16 + ml) * 64 + kk2 * 32 + koffB[j]];
        s[jn] = wmma_bf16(qf[kk2], bf_, s[jn]);
      }
    }

    // online softmax: row r of this lane is q-row (r + 8*hh)
    float p[4][8];
    __bf16* Pw = sP + wv * 16 * 64;
#pragma unroll
    for (int r = 0; r < 8; ++r) {
      float vmx = fmaxf(fmaxf(s[0][r], s[1][r]), fmaxf(s[2][r], s[3][r]));
#pragma unroll
      for (int off = 8; off > 0; off >>= 1)
        vmx = fmaxf(vmx, __shfl_xor(vmx, off, 16));
      const float mnew  = fmaxf(mstate[r], vmx);
      const float alpha = exp2f(mstate[r] - mnew);
      mstate[r] = mnew;
      float rs = 0.f;
#pragma unroll
      for (int jn = 0; jn < 4; ++jn) {
        float pv = exp2f(s[jn][r] - mnew);
        p[jn][r] = pv;
        rs += pv;
      }
#pragma unroll
      for (int off = 8; off > 0; off >>= 1) rs += __shfl_xor(rs, off, 16);
      lstate[r] = lstate[r] * alpha + rs;
#pragma unroll
      for (int jd = 0; jd < 4; ++jd) o[jd][r] = o[jd][r] * alpha;
    }
    // P: C-layout -> LDS row-major [m][key]
#pragma unroll
    for (int jn = 0; jn < 4; ++jn)
#pragma unroll
      for (int r = 0; r < 8; ++r)
        Pw[(r + 8 * hh) * 64 + jn * 16 + ml] = (__bf16)p[jn][r];
    __syncthreads();

    // O += P * V
    Frag pa[2];
#pragma unroll
    for (int kk2 = 0; kk2 < 2; ++kk2)
#pragma unroll
      for (int j = 0; j < 8; ++j)
        pa[kk2].u[j] = *(const unsigned*)&Pw[ml * 64 + kk2 * 32 + koffA[j]];
#pragma unroll
    for (int jd = 0; jd < 4; ++jd) {
#pragma unroll
      for (int kk2 = 0; kk2 < 2; ++kk2) {
        Frag bv_;
#pragma unroll
        for (int j = 0; j < 8; ++j)
          bv_.u[j] =
              *(const unsigned*)&sVT[(jd * 16 + ml) * 64 + kk2 * 32 + koffB[j]];
        o[jd] = wmma_bf16(pa[kk2], bv_, o[jd]);
      }
    }
  }

  // finalize: O /= l, write merged-head bf16 [B, L, D]
#pragma unroll
  for (int r = 0; r < 8; ++r) {
    const float inv = 1.0f / lstate[r];
    const int lq = q0 + r + 8 * hh;
#pragma unroll
    for (int jd = 0; jd < 4; ++jd) {
      const size_t idx =
          ((size_t)(bI * L_) + lq) * D_ + head * DK_ + jd * 16 + ml;
      ob[idx] = (__bf16)(o[jd][r] * inv);
    }
  }
}

// ---------------------------------------------------------------------------
extern "C" void kernel_launch(void* const* d_in, const int* in_sizes, int n_in,
                              void* d_out, int out_size, void* d_ws, size_t ws_size,
                              hipStream_t stream) {
  const float* x     = (const float*)d_in[0];
  const float* w_qkv = (const float*)d_in[1];
  const float* b_qkv = (const float*)d_in[2];
  const float* w_fc  = (const float*)d_in[3];
  const float* b_fc  = (const float*)d_in[4];
  const float* ln_g  = (const float*)d_in[5];
  const float* ln_b  = (const float*)d_in[6];
  float* out = (float*)d_out;

  // workspace carve-up (bf16), ~88 MB total
  __bf16* xn    = (__bf16*)d_ws;                       // [8192,1024]
  __bf16* wqkvb = xn    + (size_t)MTOK * D_;           // [1024,3072]
  __bf16* wfcb  = wqkvb + (size_t)D_ * N_QKV;          // [1024,1024]
  __bf16* qbuf  = wfcb  + (size_t)D_ * D_;             // [B,H,L,64]
  __bf16* kbuf  = qbuf  + (size_t)B_ * H_ * L_ * DK_;
  __bf16* vbuf  = kbuf  + (size_t)B_ * H_ * L_ * DK_;
  __bf16* abuf  = vbuf  + (size_t)B_ * H_ * L_ * DK_;  // [8192,1024]

  cvt_f32_bf16_x4<<<(D_ * N_QKV / 4 + 255) / 256, 256, 0, stream>>>(
      w_qkv, wqkvb, D_ * N_QKV / 4);
  cvt_f32_bf16_x4<<<(D_ * D_ / 4 + 255) / 256, 256, 0, stream>>>(
      w_fc, wfcb, D_ * D_ / 4);
  layernorm_bf16<<<MTOK, 256, 0, stream>>>(x, ln_g, ln_b, xn);

  gemm_bf16_kernel<<<dim3(N_QKV / 128, MTOK / 128), 256, 0, stream>>>(
      xn, wqkvb, b_qkv, MTOK, N_QKV, D_, /*mode=*/0,
      qbuf, kbuf, vbuf, nullptr, nullptr);

  attn_kernel<<<dim3(L_ / 64, B_ * H_), 128, 0, stream>>>(qbuf, kbuf, vbuf, abuf);

  gemm_bf16_kernel<<<dim3(D_ / 128, MTOK / 128), 256, 0, stream>>>(
      abuf, wfcb, b_fc, MTOK, D_, D_, /*mode=*/1,
      nullptr, nullptr, nullptr, x, out);
}
